// Prediction_32487132627603
// MI455X (gfx1250) — matrix operand order repaired and checked by reading, not compiled, not run on hardware
//
#include <hip/hip_runtime.h>
#include <hip/hip_bf16.h>
#include <math.h>

// Problem constants (from reference): B,S,H,R,IN,OPS,NPAD,L
#define B_    128
#define S_    128
#define H_    512
#define R_    16
#define IN_   2
#define OPS_  5
#define NPAD_ 16
#define L_    18
#define NEGV  (-1.0e12f)

typedef __attribute__((ext_vector_type(16))) __bf16 bf16x16;
typedef __attribute__((ext_vector_type(8)))  __bf16 bf16x8;
typedef __attribute__((ext_vector_type(8)))  float  floatx8;

__device__ __forceinline__ float sigmoidf_(float x) { return 1.0f / (1.0f + __expf(-x)); }

// ---------------------------------------------------------------------------
// Generic GEMM: C(M,N) = A(M,K) @ B(K,N), fp32 in memory, bf16 WMMA compute,
// fp32 accumulate. Workgroup = 256 threads = 8 wave32 waves, 64x64 tile.
// Wave (wm in 0..3, wn in 0..1) computes rows [wm*16,+16) x cols [wn*32,+32).
// N must be a multiple of 64, K a multiple of 32; M may be ragged.
// ---------------------------------------------------------------------------
__global__ __launch_bounds__(256)
void gemm_bf16_wmma(const float* __restrict__ A, const float* __restrict__ Bm,
                    float* __restrict__ C, int M, int K, int N) {
  __shared__ __attribute__((aligned(16))) __bf16 As[64 * 32];   // [m][k]
  __shared__ __attribute__((aligned(16))) __bf16 Bs[64 * 32];   // transposed: [n][k]

  const int tid  = threadIdx.x;
  const int n0   = blockIdx.x * 64;
  const int m0   = blockIdx.y * 64;
  const int lane = tid & 31;
  const int w    = tid >> 5;
  const int wm   = w >> 1;        // 0..3
  const int wn   = w & 1;         // 0..1
  const int half = lane >> 4;     // lane-half selects K group
  const int l16  = lane & 15;     // M row (A) / N col (B,D)

  floatx8 acc0 = {};
  floatx8 acc1 = {};

  // A staging: thread -> (row, 8-wide K octet); 2x global_load_b128, 1x ds_store_b128
  const int arow = tid >> 2;          // 0..63
  const int acol = (tid & 3) * 8;     // 0,8,16,24
  // B staging: thread -> (col n, 8-wide K octet); 8 coalesced b32 loads, 1x ds_store_b128
  const int bn   = tid & 63;          // 0..63
  const int bkq  = (tid >> 6) * 8;    // 0,8,16,24

  const bool arow_ok = (m0 + arow) < M;

  for (int k0 = 0; k0 < K; k0 += 32) {
    __syncthreads();
    // ---- Stage A tile 64x32 (f32 -> bf16), zero-pad ragged M ----
    {
      float4 a0 = make_float4(0.f, 0.f, 0.f, 0.f);
      float4 a1 = a0;
      if (arow_ok) {
        const float4* s4 = (const float4*)(A + (size_t)(m0 + arow) * K + (k0 + acol));
        a0 = s4[0];
        a1 = s4[1];
        if (k0 + 32 < K) __builtin_prefetch(A + (size_t)(m0 + arow) * K + (k0 + 32 + acol), 0, 1);
      }
      bf16x8 pk;
      pk[0] = (__bf16)a0.x; pk[1] = (__bf16)a0.y; pk[2] = (__bf16)a0.z; pk[3] = (__bf16)a0.w;
      pk[4] = (__bf16)a1.x; pk[5] = (__bf16)a1.y; pk[6] = (__bf16)a1.z; pk[7] = (__bf16)a1.w;
      *(bf16x8*)&As[arow * 32 + acol] = pk;
    }
    // ---- Stage B tile 32x64, transposed into Bs[n][k] ----
    {
      const float* src = Bm + (size_t)(k0 + bkq) * N + (n0 + bn);
      float bv[8];
#pragma unroll
      for (int j = 0; j < 8; ++j) bv[j] = src[(size_t)j * N];
      if (k0 + 32 < K) __builtin_prefetch(Bm + (size_t)(k0 + 32 + bkq) * N + (n0 + bn), 0, 1);
      bf16x8 pk;
#pragma unroll
      for (int j = 0; j < 8; ++j) pk[j] = (__bf16)bv[j];
      *(bf16x8*)&Bs[bn * 32 + bkq] = pk;
    }
    __syncthreads();

    // A fragment (ISA 16-bit A 16x32 layout): lane l16 = row M,
    // half=0 -> K 0..7 & 16..23 ; half=1 -> K 8..15 & 24..31
    bf16x16 af;
    {
      const __bf16* ar = &As[(wm * 16 + l16) * 32 + half * 8];
#pragma unroll
      for (int e = 0; e < 8; ++e) af[e] = ar[e];
#pragma unroll
      for (int e = 0; e < 8; ++e) af[8 + e] = ar[16 + e];
    }
    // B fragments (ISA B 32x16 layout): lane l16 = col N,
    // half=0 -> K 0..7 & 16..23 ; half=1 -> K 8..15 & 24..31
    bf16x16 bf0, bf1;
    {
      const __bf16* br0 = &Bs[(wn * 32 + l16) * 32 + half * 8];
      const __bf16* br1 = &Bs[(wn * 32 + 16 + l16) * 32 + half * 8];
#pragma unroll
      for (int e = 0; e < 16; ++e) bf0[e] = br0[e < 8 ? e : e + 8];
#pragma unroll
      for (int e = 0; e < 16; ++e) bf1[e] = br1[e < 8 ? e : e + 8];
    }

    acc0 = __builtin_amdgcn_wmma_f32_16x16x32_bf16(false, af, false, bf0,
                                                   (short)0, acc0, false, false);
    acc1 = __builtin_amdgcn_wmma_f32_16x16x32_bf16(false, af, false, bf1,
                                                   (short)0, acc1, false, false);
  }

  // D layout: VGPR v -> row M = v + 8*half, col = l16
#pragma unroll
  for (int v = 0; v < 8; ++v) {
    const int gm = m0 + wm * 16 + half * 8 + v;
    if (gm < M) {
      float* crow = C + (size_t)gm * N + n0 + wn * 32 + l16;
      crow[0]  = acc0[v];
      crow[16] = acc1[v];
    }
  }
}

// ---------------------------------------------------------------------------
// c = where(stack_empty, padding_hidden, current_emb); leftc = [left_childs, c]
// ---------------------------------------------------------------------------
__global__ void k_prep(const float* __restrict__ cur, const float* __restrict__ left,
                       const float* __restrict__ pad, const int* __restrict__ se,
                       float* __restrict__ c, float* __restrict__ leftc) {
  int i = blockIdx.x * blockDim.x + threadIdx.x;
  if (i >= B_ * H_) return;
  int b = i / H_, h = i % H_;
  float cv = se[b] ? pad[h] : cur[i];
  c[i] = cv;
  leftc[b * (2 * H_) + h] = left[i];
  leftc[b * (2 * H_) + H_ + h] = cv;
}

// q = where(has_left, tanh(X3+br)*sig(X4+brg), tanh(X1+bl)*sig(X2+blg))
__global__ void k_node(const float* __restrict__ X1, const float* __restrict__ X2,
                       const float* __restrict__ X3, const float* __restrict__ X4,
                       const float* __restrict__ bl, const float* __restrict__ blg,
                       const float* __restrict__ br, const float* __restrict__ brg,
                       const int* __restrict__ hl,
                       float* __restrict__ q, float* __restrict__ node_out) {
  int i = blockIdx.x * blockDim.x + threadIdx.x;
  if (i >= B_ * H_) return;
  int b = i / H_, h = i % H_;
  float nl = tanhf(X1[i] + bl[h]) * sigmoidf_(X2[i] + blg[h]);
  float nr = tanhf(X3[i] + br[h]) * sigmoidf_(X4[i] + brg[h]);
  float qv = hl[b] ? nr : nl;
  q[i] = qv;
  node_out[i] = qv;
}

// ae[s] = sum_h tanh(E_tree[s,b,h]+qW[b,h]+bta[h])*wts[h] + bts; mask; softmax_s
__global__ __launch_bounds__(128)
void k_tree_attn(const float* __restrict__ Et, const float* __restrict__ qW,
                 const float* __restrict__ bta, const float* __restrict__ wts,
                 const float* __restrict__ bts, const int* __restrict__ sm,
                 float* __restrict__ attn) {
  int b = blockIdx.x, s = threadIdx.x;
  const float* e  = Et + ((size_t)s * B_ + b) * H_;
  const float* qw = qW + b * H_;
  float acc = 0.f;
  for (int h = 0; h < H_; ++h) acc += tanhf(e[h] + qw[h] + bta[h]) * wts[h];
  float ae = acc + bts[0];
  if (sm[b * S_ + s]) ae = NEGV;
  __shared__ float sh[128];
  sh[s] = ae; __syncthreads();
  for (int st = 64; st > 0; st >>= 1) { if (s < st) sh[s] = fmaxf(sh[s], sh[s + st]); __syncthreads(); }
  float mx = sh[0]; __syncthreads();
  float ex = __expf(ae - mx);
  sh[s] = ex; __syncthreads();
  for (int st = 64; st > 0; st >>= 1) { if (s < st) sh[s] += sh[s + st]; __syncthreads(); }
  attn[b * S_ + s] = ex / sh[0];
}

// current_context[b,h] = sum_s attn[b,s]*enc[s,b,h]; leaf = [q, ctx]
__global__ void k_ctx(const float* __restrict__ attn, const float* __restrict__ enc,
                      const float* __restrict__ q,
                      float* __restrict__ ctx_out, float* __restrict__ leaf) {
  int i = blockIdx.x * blockDim.x + threadIdx.x;
  if (i >= B_ * H_) return;
  int b = i / H_, h = i % H_;
  const float* a = attn + b * S_;
  float acc = 0.f;
  for (int s = 0; s < S_; ++s) acc += a[s] * enc[((size_t)s * B_ + b) * H_ + h];
  ctx_out[i] = acc;
  leaf[b * (2 * H_) + h] = q[i];
  leaf[b * (2 * H_) + H_ + h] = acc;
}

// per (b,r): score[s] = sum_h tanh(E_rule[s,b,h]+Rr[r,h]+bra[h])*wrs[h] + brs;
// mask; softmax_s; comb = prob + 0.2*current_attn
__global__ __launch_bounds__(128)
void k_rule_attn(const float* __restrict__ Er, const float* __restrict__ Rr,
                 const float* __restrict__ bra, const float* __restrict__ wrs,
                 const float* __restrict__ brs, const int* __restrict__ sm,
                 const float* __restrict__ attn, float* __restrict__ comb) {
  int br_ = blockIdx.x;
  int b = br_ / R_, r = br_ % R_;
  int s = threadIdx.x;
  const float* e  = Er + ((size_t)s * B_ + b) * H_;
  const float* rr = Rr + r * H_;
  float acc = 0.f;
  for (int h = 0; h < H_; ++h) acc += tanhf(e[h] + rr[h] + bra[h]) * wrs[h];
  float ae = acc + brs[0];
  if (sm[b * S_ + s]) ae = NEGV;
  __shared__ float sh[128];
  sh[s] = ae; __syncthreads();
  for (int st = 64; st > 0; st >>= 1) { if (s < st) sh[s] = fmaxf(sh[s], sh[s + st]); __syncthreads(); }
  float mx = sh[0]; __syncthreads();
  float ex = __expf(ae - mx);
  sh[s] = ex; __syncthreads();
  for (int st = 64; st > 0; st >>= 1) { if (s < st) sh[s] += sh[s + st]; __syncthreads(); }
  comb[((size_t)b * R_ + r) * S_ + s] = ex / sh[0] + 0.2f * attn[b * S_ + s];
}

// rules_context[b,r,h] = sum_s comb[b,r,s]*enc[s,b,h]
__global__ void k_rules_ctx(const float* __restrict__ comb, const float* __restrict__ enc,
                            float* __restrict__ rc) {
  int i = blockIdx.x * blockDim.x + threadIdx.x;
  if (i >= B_ * R_ * H_) return;
  int b = i / (R_ * H_);
  int r = (i / H_) % R_;
  int h = i % H_;
  const float* cb = comb + ((size_t)b * R_ + r) * S_;
  float acc = 0.f;
  for (int s = 0; s < S_; ++s) acc += cb[s] * enc[((size_t)s * B_ + b) * H_ + h];
  rc[i] = acc;
}

// rule_prob[b,:] = softmax_r( sum_h tanh(RP[b,r,h]+brp1[h])*wrp2[h] + brp2 )
__global__ __launch_bounds__(128)
void k_rule_prob(const float* __restrict__ RP, const float* __restrict__ brp1,
                 const float* __restrict__ wrp2, const float* __restrict__ brp2,
                 float* __restrict__ out) {
  int b = blockIdx.x, t = threadIdx.x;
  int r = t >> 3, p = t & 7;          // 16 rules x 8 partials
  const float* row = RP + ((size_t)b * R_ + r) * H_;
  float acc = 0.f;
  for (int h = p * 64; h < p * 64 + 64; ++h) acc += tanhf(row[h] + brp1[h]) * wrp2[h];
  __shared__ float sh[128];
  __shared__ float sc[R_ + 2];
  sh[t] = acc; __syncthreads();
  if (t < R_) {
    float sum = 0.f;
    for (int j = 0; j < 8; ++j) sum += sh[t * 8 + j];
    sc[t] = sum + brp2[0];
  }
  __syncthreads();
  if (t == 0) {
    float mx = sc[0];
    for (int j = 1; j < R_; ++j) mx = fmaxf(mx, sc[j]);
    float sum = 0.f;
    for (int j = 0; j < R_; ++j) sum += __expf(sc[j] - mx);
    sc[R_] = mx; sc[R_ + 1] = sum;
  }
  __syncthreads();
  if (t < R_) out[b * R_ + t] = __expf(sc[t] - sc[R_]) / sc[R_ + 1];
}

// ew = concat(broadcast(embedding_weight), num_pades)  -> (B, L, H)
__global__ void k_ew(const float* __restrict__ embw, const float* __restrict__ npad,
                     float* __restrict__ ew) {
  int i = blockIdx.x * blockDim.x + threadIdx.x;
  if (i >= B_ * L_ * H_) return;
  int b = i / (L_ * H_);
  int n = (i / H_) % L_;
  int h = i % H_;
  ew[i] = (n < IN_) ? embw[n * H_ + h] : npad[((size_t)b * NPAD_ + (n - IN_)) * H_ + h];
}

// num_score[b,n] = mask ? NEG : sum_h tanh(leafW[b,h]+ewW[b,n,h]+bsc[h])*wss[h]
__global__ __launch_bounds__(128)
void k_numscore(const float* __restrict__ leafW, const float* __restrict__ ewW,
                const float* __restrict__ bsc, const float* __restrict__ wss,
                const int* __restrict__ nm, float* __restrict__ out) {
  int bn = blockIdx.x;
  int b = bn / L_;
  int t = threadIdx.x;
  const float* lw = leafW + b * H_;
  const float* ew = ewW + (size_t)bn * H_;
  float acc = 0.f;
  for (int h = t; h < H_; h += 128) acc += tanhf(lw[h] + ew[h] + bsc[h]) * wss[h];
  __shared__ float sh[128];
  sh[t] = acc; __syncthreads();
  for (int st = 64; st > 0; st >>= 1) { if (t < st) sh[t] += sh[t + st]; __syncthreads(); }
  if (t == 0) out[bn] = nm[bn] ? NEGV : sh[0];
}

// op = leaf @ W_ops + b_ops   (tiny)
__global__ void k_op(const float* __restrict__ leaf, const float* __restrict__ Wops,
                     const float* __restrict__ bops, float* __restrict__ out) {
  int i = blockIdx.x * blockDim.x + threadIdx.x;
  if (i >= B_ * OPS_) return;
  int b = i / OPS_, o = i % OPS_;
  float acc = bops[o];
  for (int k = 0; k < 2 * H_; ++k) acc += leaf[b * 2 * H_ + k] * Wops[k * OPS_ + o];
  out[i] = acc;
}

// ---------------------------------------------------------------------------
extern "C" void kernel_launch(void* const* d_in, const int* in_sizes, int n_in,
                              void* d_out, int out_size, void* d_ws, size_t ws_size,
                              hipStream_t stream) {
  (void)in_sizes; (void)n_in; (void)out_size; (void)ws_size;

  const float* enc  = (const float*)d_in[0];
  const float* cur  = (const float*)d_in[1];
  const float* left = (const float*)d_in[2];
  const float* pad  = (const float*)d_in[3];
  const float* npad = (const float*)d_in[4];
  const float* rsc  = (const float*)d_in[5];
  const float* embw = (const float*)d_in[6];
  const float* Wl   = (const float*)d_in[7];
  const float* bl   = (const float*)d_in[8];
  const float* Wlg  = (const float*)d_in[9];
  const float* blg  = (const float*)d_in[10];
  const float* Wr   = (const float*)d_in[11];
  const float* br   = (const float*)d_in[12];
  const float* Wrg  = (const float*)d_in[13];
  const float* brg  = (const float*)d_in[14];
  const float* Wta  = (const float*)d_in[15];
  const float* bta  = (const float*)d_in[16];
  const float* Wts  = (const float*)d_in[17];
  const float* bts  = (const float*)d_in[18];
  const float* Wra  = (const float*)d_in[19];
  const float* bra  = (const float*)d_in[20];
  const float* Wrsc = (const float*)d_in[21];
  const float* brsc = (const float*)d_in[22];
  const float* Wrp1 = (const float*)d_in[23];
  const float* brp1 = (const float*)d_in[24];
  const float* Wrp2 = (const float*)d_in[25];
  const float* brp2 = (const float*)d_in[26];
  const float* Wsa  = (const float*)d_in[27];
  const float* bsa  = (const float*)d_in[28];
  const float* Wss  = (const float*)d_in[29];
  const float* Wops = (const float*)d_in[30];
  const float* bops = (const float*)d_in[31];
  const int* sm = (const int*)d_in[32];
  const int* nm = (const int*)d_in[33];
  const int* hl = (const int*)d_in[34];
  const int* se = (const int*)d_in[35];

  // Output layout (floats, concatenated in reference return order)
  float* out    = (float*)d_out;
  float* o_num  = out;             // (B,L)        2304
  float* o_op   = out + 2304;      // (B,OPS)      640
  float* o_node = out + 2944;      // (B,1,H)      65536
  float* o_ctx  = out + 68480;     // (B,1,H)      65536
  float* o_ew   = out + 134016;    // (B,L,H)      1179648
  float* o_rpb  = out + 1313664;   // (B,R)        2048

  // Workspace layout
  float* ws     = (float*)d_ws;
  float* w_c    = ws;                  // B*H
  float* w_lc   = w_c    + 65536;      // B*2H
  float* w_x1   = w_lc   + 131072;     // B*H
  float* w_x2   = w_x1   + 65536;
  float* w_x3   = w_x2   + 65536;
  float* w_x4   = w_x3   + 65536;
  float* w_q    = w_x4   + 65536;      // B*H
  float* w_qw   = w_q    + 65536;      // B*H
  float* w_rr   = w_qw   + 65536;      // R*H
  float* w_attn = w_rr   + 8192;       // B*S
  float* w_leaf = w_attn + 16384;      // B*2H
  float* w_comb = w_leaf + 131072;     // B*R*S
  float* w_rc   = w_comb + 262144;     // B*R*H
  float* w_rp   = w_rc   + 1048576;    // B*R*H
  float* w_lw   = w_rp   + 1048576;    // B*H
  float* w_eww  = w_lw   + 65536;      // B*L*H
  float* w_big  = w_eww  + 1179648;    // S*B*H  (E_tree then E_rule)

  auto gemm = [&](const float* A, const float* Bmat, float* C, int M, int K, int N) {
    dim3 grid(N / 64, (M + 63) / 64);
    gemm_bf16_wmma<<<grid, dim3(256), 0, stream>>>(A, Bmat, C, M, K, N);
  };

  const int nBH = B_ * H_;

  // 1) gated node embeddings
  k_prep<<<(nBH + 255) / 256, 256, 0, stream>>>(cur, left, pad, se, w_c, w_lc);
  gemm(w_c,  Wl,  w_x1, B_, H_,     H_);
  gemm(w_c,  Wlg, w_x2, B_, H_,     H_);
  gemm(w_lc, Wr,  w_x3, B_, 2 * H_, H_);
  gemm(w_lc, Wrg, w_x4, B_, 2 * H_, H_);
  k_node<<<(nBH + 255) / 256, 256, 0, stream>>>(w_x1, w_x2, w_x3, w_x4,
                                                bl, blg, br, brg, hl, w_q, o_node);

  // 2) tree attention
  gemm(w_q, Wta,            w_qw,  B_,      H_, H_);   // q @ W_tree_attn[:h]
  gemm(enc, Wta + H_ * H_,  w_big, S_ * B_, H_, H_);   // enc @ W_tree_attn[h:]
  k_tree_attn<<<B_, 128, 0, stream>>>(w_big, w_qw, bta, Wts, bts, sm, w_attn);
  k_ctx<<<(nBH + 255) / 256, 256, 0, stream>>>(w_attn, enc, w_q, o_ctx, w_leaf);

  // 3) rule attention + rule prob
  gemm(rsc, Wra + H_ * H_, w_rr,  R_,      H_, H_);    // rule_scores @ W_rule_attn[h:]
  gemm(enc, Wra,           w_big, S_ * B_, H_, H_);    // enc @ W_rule_attn[:h] (reuse big)
  k_rule_attn<<<B_ * R_, 128, 0, stream>>>(w_big, w_rr, bra, Wrsc, brsc, sm, w_attn, w_comb);
  k_rules_ctx<<<(B_ * R_ * H_ + 255) / 256, 256, 0, stream>>>(w_comb, enc, w_rc);
  gemm(w_rc, Wrp1, w_rp, B_ * R_, H_, H_);
  k_rule_prob<<<B_, 128, 0, stream>>>(w_rp, brp1, Wrp2, brp2, o_rpb);

  // 4) number scores + ops
  k_ew<<<(B_ * L_ * H_ + 255) / 256, 256, 0, stream>>>(embw, npad, o_ew);
  gemm(w_leaf, Wsa,               w_lw,  B_,      2 * H_, H_);  // leaf @ W_sc_attn[:2h]
  gemm(o_ew,   Wsa + 2 * H_ * H_, w_eww, B_ * L_, H_,     H_);  // ew @ W_sc_attn[2h:]
  k_numscore<<<B_ * L_, 128, 0, stream>>>(w_lw, w_eww, bsa, Wss, nm, o_num);
  k_op<<<(B_ * OPS_ + 255) / 256, 256, 0, stream>>>(w_leaf, Wops, bops, o_op);
}
